// TinyTransformBlock_4844723109982
// MI455X (gfx1250) — matrix operand order repaired and checked by reading, not compiled, not run on hardware
//
#include <hip/hip_runtime.h>
#include <stdint.h>

// ---------------- problem constants ----------------
#define L_LAYERS 4
#define DIM      1024
#define NHEAD    16
#define DHEAD    64
#define INNER    1024   // NHEAD*DHEAD
#define MUL      4
#define HID      4096   // DIM*MUL
#define BATCH    2
#define SEQ      2048
#define TOK      4096   // BATCH*SEQ

typedef __attribute__((ext_vector_type(16))) _Float16 v16h;
typedef __attribute__((ext_vector_type(8)))  _Float16 v8h;
typedef __attribute__((ext_vector_type(8)))  float    v8f;
typedef __attribute__((ext_vector_type(4)))  int      v4i;

static __device__ inline v16h cat8(v8h lo, v8h hi) {
  v16h r;
#pragma unroll
  for (int i = 0; i < 8; ++i) { r[i] = lo[i]; r[i + 8] = hi[i]; }
  return r;
}
// 16 contiguous halves (B fragment / V fragment): k = k0 + j
static __device__ inline v16h load16c(const _Float16* p) {
  const v8h* q = (const v8h*)p;
  return cat8(q[0], q[1]);
}
// A fragment: 16 contiguous halves [p..p+15]
static __device__ inline v16h load_afrag(const _Float16* p) {
  const v8h* q = (const v8h*)p;
  return cat8(q[0], q[1]);
}
static __device__ inline v8f wmma_f16(v16h a, v16h b, v8f c) {
  return __builtin_amdgcn_wmma_f32_16x16x32_f16(false, a, false, b, (short)0, c,
                                                false, false);
}
static __device__ inline void wave_lds_fence() {
  asm volatile("s_wait_dscnt 0" ::: "memory");
}
static __device__ inline void async_fence() {
  asm volatile("s_wait_asynccnt 0" ::: "memory");
}
// 16-byte async global->LDS copy (f16 data already in device layout).
// Builtin signature (probed): (v4i as1* src, v4i as3* dst, imm off, imm cpol).
// as1 pointer == flat-global bit pattern; as3 offset == low 32 bits of a flat
// shared address (LDS aperture truncation rule).
static __device__ inline void async_copy_b128(const _Float16* g, _Float16* l) {
  __attribute__((address_space(1))) v4i* gp =
      (__attribute__((address_space(1))) v4i*)(uint64_t)(uintptr_t)g;
  __attribute__((address_space(3))) v4i* lp =
      (__attribute__((address_space(3))) v4i*)(uint32_t)(uintptr_t)l;
  __builtin_amdgcn_global_load_async_to_lds_b128(gp, lp, 0, 0);
}

// ---------------- layernorm: fp32 row -> f16 normalized row ----------------
__global__ __launch_bounds__(256) void ln_kernel(const float* __restrict__ x,
                                                 const float* __restrict__ g,
                                                 const float* __restrict__ b,
                                                 _Float16* __restrict__ out) {
  const int row = blockIdx.x;
  const float* xr = x + (size_t)row * DIM;
  float s = 0.f, s2 = 0.f;
  for (int c = threadIdx.x; c < DIM; c += 256) {
    float v = xr[c];
    s += v; s2 += v * v;
  }
  __shared__ float r1[256], r2[256];
  r1[threadIdx.x] = s; r2[threadIdx.x] = s2;
  __syncthreads();
  for (int off = 128; off > 0; off >>= 1) {
    if (threadIdx.x < off) {
      r1[threadIdx.x] += r1[threadIdx.x + off];
      r2[threadIdx.x] += r2[threadIdx.x + off];
    }
    __syncthreads();
  }
  const float mu  = r1[0] * (1.f / DIM);
  const float var = r2[0] * (1.f / DIM) - mu * mu;
  const float rs  = rsqrtf(var + 1e-5f);
  for (int c = threadIdx.x; c < DIM; c += 256) {
    float v = (xr[c] - mu) * rs * g[c] + b[c];
    out[(size_t)row * DIM + c] = (_Float16)v;
  }
}

// ---------------- generic WMMA GEMM: C = A(f16,MxK) * B(f32,KxN) ------------
// mode 0: store f16            (QKV projections)
// mode 1: +bias, GELU, f16     (fc1)
// mode 2: +bias, residual add into fp32 out (Wo proj, fc2)
__global__ __launch_bounds__(256) void gemm_kernel(
    const _Float16* __restrict__ A, int lda, const float* __restrict__ Bw,
    int ldb, int M, int N, int K, const float* __restrict__ bias, int mode,
    float* __restrict__ outF, _Float16* __restrict__ outH) {
  __shared__ _Float16 sa[128][40];  // [m][k], padded
  __shared__ _Float16 sb[128][40];  // [n][k] (B transposed), padded

  const int t = threadIdx.x;
  const int lane = t & 31, wid = t >> 5;
  const int l15 = lane & 15, lhi = lane >> 4;
  const int m0 = blockIdx.y * 128, n0 = blockIdx.x * 128;
  const int rowBase = (wid >> 2) * 64;   // 0 or 64
  const int colBase = (wid & 3) * 32;    // 0,32,64,96

  v8f acc[4][2];
  const v8f zf = {0.f, 0.f, 0.f, 0.f, 0.f, 0.f, 0.f, 0.f};
#pragma unroll
  for (int i = 0; i < 4; ++i)
#pragma unroll
    for (int j = 0; j < 2; ++j) acc[i][j] = zf;

  const int ar = t >> 1, ac = (t & 1) * 16;        // A-tile copy coords
  const int bn = t & 127, bk0 = (t >> 7) * 16;     // B-tile copy coords

  for (int kb = 0; kb < K; kb += 32) {
    __syncthreads();
    {  // stage A tile: async f16 global -> LDS (no VGPR round trip)
      const _Float16* src = A + (size_t)(m0 + ar) * lda + kb + ac;
      _Float16* dst = &sa[ar][ac];
      async_copy_b128(src, dst);
      async_copy_b128(src + 8, dst + 8);
    }
    {  // stage B tile transposed (f32 global -> f16 LDS), convert on the fly
      const float* src = Bw + (size_t)(kb + bk0) * ldb + (n0 + bn);
#pragma unroll
      for (int j = 0; j < 16; ++j)
        sb[bn][bk0 + j] = (_Float16)src[(size_t)j * ldb];
    }
    async_fence();
    __syncthreads();

    v16h af[4];
#pragma unroll
    for (int mr = 0; mr < 4; ++mr)
      af[mr] = load_afrag(&sa[rowBase + mr * 16 + l15][lhi * 16]);
#pragma unroll
    for (int nr = 0; nr < 2; ++nr) {
      v16h bf = load16c(&sb[colBase + nr * 16 + l15][lhi * 16]);
#pragma unroll
      for (int mr = 0; mr < 4; ++mr)
        acc[mr][nr] = wmma_f16(af[mr], bf, acc[mr][nr]);
    }
  }

  // epilogue (C layout: lane holds col l15, rows lhi*8 + r)
#pragma unroll
  for (int mr = 0; mr < 4; ++mr)
#pragma unroll
    for (int nr = 0; nr < 2; ++nr) {
      const int gm0 = m0 + rowBase + mr * 16 + lhi * 8;
      const int gn  = n0 + colBase + nr * 16 + l15;
      const float bv = bias ? bias[gn] : 0.f;
#pragma unroll
      for (int r = 0; r < 8; ++r) {
        float v = acc[mr][nr][r] + bv;
        const size_t idx = (size_t)(gm0 + r) * N + gn;
        if (mode == 1) {
          v = 0.5f * v * (1.f + erff(v * 0.70710678118654752f));
          outH[idx] = (_Float16)v;
        } else if (mode == 2) {
          outF[idx] += v;
        } else {
          outH[idx] = (_Float16)v;
        }
      }
    }
}

// ---------------- flash attention (causal) with WMMA ------------------------
// grid: (SEQ/64, BATCH*NHEAD), block: 128 (4 waves, 16 query rows each)
__global__ __launch_bounds__(128) void attn_kernel(
    const _Float16* __restrict__ Q, const _Float16* __restrict__ K,
    const _Float16* __restrict__ V, _Float16* __restrict__ O) {
  const int qblk = blockIdx.x;
  const int bh = blockIdx.y;
  const int b = bh >> 4, h = bh & 15;
  const int wid = threadIdx.x >> 5, lane = threadIdx.x & 31;
  const int l15 = lane & 15, lhi = lane >> 4;
  const int qRow0 = qblk * 64 + wid * 16;  // this wave's first query row
  const size_t tokBase = (size_t)b * SEQ;

  __shared__ _Float16 svT[64][32];      // V^T tile [d][key], shared by block
  __shared__ float    sS[4][16][33];    // per-wave scores
  __shared__ _Float16 sP[4][16][32];    // per-wave probabilities (A-frag src)
  __shared__ float    sScale[4][16];
  __shared__ float    sLsum[4][16];

  // Q fragments (16 rows x 64 d = two 16x32 A-fragments), loaded once
  const _Float16* qbase =
      Q + (tokBase + qRow0 + l15) * INNER + h * DHEAD + lhi * 16;
  v16h qf[2];
  qf[0] = load_afrag(qbase);
  qf[1] = load_afrag(qbase + 32);

  v8f acc[4];
  const v8f zf = {0.f, 0.f, 0.f, 0.f, 0.f, 0.f, 0.f, 0.f};
#pragma unroll
  for (int i = 0; i < 4; ++i) acc[i] = zf;
  float mrun = -1e30f, lsum = 0.f;  // stats live in lanes 0..15 (row = lane)

  const int nkt = qblk * 2 + 2;     // 32-key tiles covering causal range
  const int waveMaxRow = qRow0 + 15;

  for (int kt = 0; kt < nkt; ++kt) {
    const int kb = kt * 32;
    __syncthreads();
    {  // stage V^T tile: each thread 16 halves of one key row
      const int key = threadIdx.x >> 2;
      const int d0 = (threadIdx.x & 3) * 16;
      const _Float16* vp = V + (tokBase + kb + key) * INNER + h * DHEAD + d0;
#pragma unroll
      for (int j = 0; j < 16; ++j) svT[d0 + j][key] = vp[j];
    }
    __syncthreads();

    if (kb <= waveMaxRow) {  // wave-uniform predicate -> EXEC all 1s inside
      // ---- scores: S = Q K^T * scale, two 16-key halves ----
#pragma unroll
      for (int half = 0; half < 2; ++half) {
        const int keyN0 = kb + half * 16;
        v8f s = zf;
#pragma unroll
        for (int c = 0; c < 2; ++c) {
          const _Float16* kp = K + (tokBase + keyN0 + l15) * INNER + h * DHEAD +
                               c * 32 + lhi * 16;
          v16h kf = load16c(kp);  // K^T B-fragment is contiguous in K storage
          s = wmma_f16(qf[c], kf, s);
        }
        const int colKey = keyN0 + l15;
#pragma unroll
        for (int r = 0; r < 8; ++r) {
          const int rowq = qRow0 + lhi * 8 + r;
          float v = s[r] * 0.125f;  // DH^-0.5
          if (colKey > rowq) v = -1e30f;
          sS[wid][lhi * 8 + r][half * 16 + l15] = v;
        }
      }
      wave_lds_fence();
      // ---- online softmax stats (lanes 0..15, one row each) ----
      if (lane < 16) {
        const int row = lane;
        float tmax = -1e30f;
#pragma unroll
        for (int c = 0; c < 32; ++c) tmax = fmaxf(tmax, sS[wid][row][c]);
        const float nm = fmaxf(mrun, tmax);
        const float sc = __expf(mrun - nm);
        float ssum = 0.f;
#pragma unroll
        for (int c = 0; c < 32; ++c) {
          float p = __expf(sS[wid][row][c] - nm);
          ssum += p;
          sP[wid][row][c] = (_Float16)p;
        }
        mrun = nm;
        lsum = lsum * sc + ssum;
        sScale[wid][row] = sc;
      }
      wave_lds_fence();
      // ---- rescale output accumulators ----
      float scv[8];
#pragma unroll
      for (int r = 0; r < 8; ++r) scv[r] = sScale[wid][lhi * 8 + r];
#pragma unroll
      for (int t4 = 0; t4 < 4; ++t4)
#pragma unroll
        for (int r = 0; r < 8; ++r) acc[t4][r] *= scv[r];
      // ---- O += P V : one P A-fragment, four V B-fragments ----
      v16h pf = load_afrag(&sP[wid][l15][lhi * 16]);
#pragma unroll
      for (int nr = 0; nr < 4; ++nr) {
        v16h vf = load16c(&svT[nr * 16 + l15][lhi * 16]);
        acc[nr] = wmma_f16(pf, vf, acc[nr]);
      }
    }
  }

  // ---- finalize: divide by row sums, store f16 ----
  if (lane < 16) sLsum[wid][lane] = lsum;
  wave_lds_fence();
  float inv[8];
#pragma unroll
  for (int r = 0; r < 8; ++r) inv[r] = 1.f / sLsum[wid][lhi * 8 + r];
#pragma unroll
  for (int nr = 0; nr < 4; ++nr)
#pragma unroll
    for (int r = 0; r < 8; ++r) {
      const size_t row = tokBase + qRow0 + lhi * 8 + r;
      O[row * INNER + h * DHEAD + nr * 16 + l15] = (_Float16)(acc[nr][r] * inv[r]);
    }
}

// ---------------- host orchestration ----------------------------------------
extern "C" void kernel_launch(void* const* d_in, const int* in_sizes, int n_in,
                              void* d_out, int out_size, void* d_ws,
                              size_t ws_size, hipStream_t stream) {
  (void)in_sizes; (void)n_in; (void)out_size; (void)ws_size;
  const float* x      = (const float*)d_in[0];
  const float* ln1_g  = (const float*)d_in[1];
  const float* ln1_b  = (const float*)d_in[2];
  const float* Wq     = (const float*)d_in[3];
  const float* Wk     = (const float*)d_in[4];
  const float* Wv     = (const float*)d_in[5];
  const float* Wo     = (const float*)d_in[6];
  const float* ln3_g  = (const float*)d_in[7];
  const float* ln3_b  = (const float*)d_in[8];
  const float* fc1_w  = (const float*)d_in[9];
  const float* fc1_b  = (const float*)d_in[10];
  const float* fc2_w  = (const float*)d_in[11];
  const float* fc2_b  = (const float*)d_in[12];
  float* xcur = (float*)d_out;  // running residual stream (fp32)

  _Float16* wsh = (_Float16*)d_ws;
  _Float16* xn = wsh;                                   // TOK*DIM
  _Float16* q  = wsh + (size_t)TOK * DIM;               // TOK*INNER
  _Float16* k  = q + (size_t)TOK * INNER;
  _Float16* v  = k + (size_t)TOK * INNER;
  _Float16* ao = v + (size_t)TOK * INNER;
  _Float16* hh = q;  // alias: q/k/v/ao dead when h (TOK*HID) is live

  (void)hipMemcpyAsync(xcur, x, (size_t)TOK * DIM * sizeof(float),
                       hipMemcpyDeviceToDevice, stream);

  const dim3 blk256(256), blk128(128);
  const dim3 gDIM(DIM / 128, TOK / 128);   // N=1024
  const dim3 gHID(HID / 128, TOK / 128);   // N=4096
  const dim3 gATT(SEQ / 64, BATCH * NHEAD);

  for (int l = 0; l < L_LAYERS; ++l) {
    // xn = LN1(x)
    ln_kernel<<<TOK, blk256, 0, stream>>>(xcur, ln1_g + (size_t)l * DIM,
                                          ln1_b + (size_t)l * DIM, xn);
    // Q/K/V = xn @ W{q,k,v}
    gemm_kernel<<<gDIM, blk256, 0, stream>>>(xn, DIM, Wq + (size_t)l * DIM * INNER,
                                             INNER, TOK, INNER, DIM, nullptr, 0,
                                             nullptr, q);
    gemm_kernel<<<gDIM, blk256, 0, stream>>>(xn, DIM, Wk + (size_t)l * DIM * INNER,
                                             INNER, TOK, INNER, DIM, nullptr, 0,
                                             nullptr, k);
    gemm_kernel<<<gDIM, blk256, 0, stream>>>(xn, DIM, Wv + (size_t)l * DIM * INNER,
                                             INNER, TOK, INNER, DIM, nullptr, 0,
                                             nullptr, v);
    // ao = causal attention(Q,K,V)
    attn_kernel<<<gATT, blk128, 0, stream>>>(q, k, v, ao);
    // x += ao @ Wo
    gemm_kernel<<<gDIM, blk256, 0, stream>>>(ao, INNER,
                                             Wo + (size_t)l * INNER * DIM, DIM,
                                             TOK, DIM, INNER, nullptr, 2, xcur,
                                             nullptr);
    // h = gelu(LN3(x) @ fc1 + b1)
    ln_kernel<<<TOK, blk256, 0, stream>>>(xcur, ln3_g + (size_t)l * DIM,
                                          ln3_b + (size_t)l * DIM, xn);
    gemm_kernel<<<gHID, blk256, 0, stream>>>(xn, DIM,
                                             fc1_w + (size_t)l * DIM * HID, HID,
                                             TOK, HID, DIM,
                                             fc1_b + (size_t)l * HID, 1, nullptr,
                                             hh);
    // x += h @ fc2 + b2
    gemm_kernel<<<gDIM, blk256, 0, stream>>>(hh, HID,
                                             fc2_w + (size_t)l * HID * DIM, DIM,
                                             TOK, DIM, HID,
                                             fc2_b + (size_t)l * DIM, 2, xcur,
                                             nullptr);
  }
}
